// DecoderRNN_23493471109562
// MI455X (gfx1250) — matrix operand order repaired and checked by reading, not compiled, not run on hardware
//
#include <hip/hip_runtime.h>
#include <hip/hip_bf16.h>
#include <math.h>

// ---------------------------------------------------------------------------
// Problem constants (match reference)
// ---------------------------------------------------------------------------
#define BB 32
#define SS 64
#define VV 32000
#define EE 512
#define HH 512
#define G3 (3 * HH)      // 1536
#define EOS_IDX 1

typedef __bf16 bf16;
typedef __attribute__((ext_vector_type(16))) __bf16 v16bf;
typedef __attribute__((ext_vector_type(8)))  float  v8f;

// ---------------------------------------------------------------------------
// WMMA helpers: D = A(16x32 bf16) x B(32x16 bf16) + C(16x16 f32)
// ---------------------------------------------------------------------------
__device__ __forceinline__ v8f wmma_bf16(v16bf a, v16bf b, v8f c) {
    return __builtin_amdgcn_wmma_f32_16x16x32_bf16(
        /*neg_a=*/false, a, /*neg_b=*/false, b,
        /*c_mod=*/(short)0, c, /*reuse_a=*/false, /*reuse_b=*/false);
}

// A is row-major (32 x K). Fragment covers rows m0..m0+15, cols k0..k0+31.
// Lane l: row = m0 + (l&15); lanes 0-15 hold K k0..k0+15,
// lanes 16-31 hold K k0+16..k0+31 (contiguous, matching load_b_frag).
__device__ __forceinline__ v16bf load_a_frag(const bf16* __restrict__ A, int lda,
                                             int m0, int k0, int lane) {
    int m  = m0 + (lane & 15);
    int hi = lane >> 4;
    const bf16* p = A + (size_t)m * lda + k0 + hi * 16;
    v16bf a;
#pragma unroll
    for (int j = 0; j < 8; ++j) {
        a[j]     = p[j];        // K = k0 + hi*16 + j
        a[j + 8] = p[8 + j];    // K = k0 + hi*16 + 8 + j
    }
    return a;
}

// W is row-major (N x K); we need B[k][n] = W[n][k] (a 32x16 K-major tile).
// Lane l: col n = n0 + (l&15); lanes 0-15 hold K k0..k0+15, lanes 16-31 hold
// K k0+16..k0+31, packed sequentially two-per-VGPR. Contiguous 32B per lane.
__device__ __forceinline__ v16bf load_b_frag(const bf16* __restrict__ W, int ldw,
                                             int n0, int k0, int lane) {
    int n  = n0 + (lane & 15);
    int hi = lane >> 4;
    const bf16* p = W + (size_t)n * ldw + k0 + hi * 16;
    v16bf b;
#pragma unroll
    for (int j = 0; j < 16; ++j) b[j] = p[j];
    return b;
}

// ---------------------------------------------------------------------------
// Generic skinny GEMM: C[32 x N] = A[32 x K](bf16) * W[N x K](bf16)^T (+bias)
// Block = 128 threads (4 waves). Each wave owns a 16-column strip and both
// 16-row M tiles, so the streamed W fragment is loaded once and reused twice.
// Grid.x = N / 64.
// ---------------------------------------------------------------------------
__global__ void __launch_bounds__(128)
gemm32_bf16_kernel(const bf16* __restrict__ A, const bf16* __restrict__ W,
                   const float* __restrict__ bias, float* __restrict__ C,
                   int N, int K, int addBias) {
    int lane = threadIdx.x & 31;
    int wave = threadIdx.x >> 5;                  // 0..3
    int n0   = blockIdx.x * 64 + wave * 16;

    v8f acc0 = {};
    v8f acc1 = {};
    for (int k0 = 0; k0 < K; k0 += 32) {
        v16bf bfrag = load_b_frag(W, K, n0, k0, lane);
        v16bf a0    = load_a_frag(A, K, 0,  k0, lane);
        v16bf a1    = load_a_frag(A, K, 16, k0, lane);
        acc0 = wmma_bf16(a0, bfrag, acc0);
        acc1 = wmma_bf16(a1, bfrag, acc1);
    }

    int n  = n0 + (lane & 15);
    int hi = lane >> 4;
    float bv = addBias ? bias[n] : 0.0f;
#pragma unroll
    for (int r = 0; r < 8; ++r) {
        int m = r + 8 * hi;                       // rows per ISA C/D layout
        C[(size_t)m        * N + n] = acc0[r] + bv;
        C[(size_t)(m + 16) * N + n] = acc1[r] + bv;
    }
}

// ---------------------------------------------------------------------------
// Setup kernels
// ---------------------------------------------------------------------------
__global__ void f32_to_bf16_kernel(const float* __restrict__ src,
                                   bf16* __restrict__ dst, int n) {
    int i = blockIdx.x * blockDim.x + threadIdx.x;
    if (i < n) dst[i] = (bf16)src[i];
}

__global__ void init_state_kernel(const float* __restrict__ sent_state,
                                  float* __restrict__ word_h, float* __restrict__ sent_h,
                                  bf16* __restrict__ word_h_bf, bf16* __restrict__ sent_h_bf,
                                  float* __restrict__ loss_out) {
    int i = blockIdx.x * blockDim.x + threadIdx.x;  // 0 .. B*H-1
    if (i < BB * HH) {
        float v = sent_state[i];                    // sent_state[0] is (B,H)
        word_h[i] = v; sent_h[i] = v;
        word_h_bf[i] = (bf16)v; sent_h_bf[i] = (bf16)v;
    }
    if (i == 0) loss_out[0] = 0.0f;
}

// x = concat(emb[kw], emb[lw]) per batch row, emitted as bf16 (32 x 1024)
__global__ void embed_kernel(const float* __restrict__ emb,
                             const int* __restrict__ targets,
                             const int* __restrict__ targets_kws,
                             int t, bf16* __restrict__ x_bf) {
    int idx = blockIdx.x * blockDim.x + threadIdx.x;    // 0 .. 32*1024-1
    int b = idx >> 10;
    int c = idx & 1023;
    int lw = targets[b * SS + t];                       // last word  (t)
    int kw = targets_kws[b * SS + t + 1];               // keyword    (t+1)
    float v = (c < EE) ? emb[(size_t)kw * EE + c]
                       : emb[(size_t)lw * EE + (c - EE)];
    x_bf[idx] = (bf16)v;
}

// ---------------------------------------------------------------------------
// GRU gate math (PyTorch order r,z,n). n-gate needs gi and gh separately.
// ---------------------------------------------------------------------------
__device__ __forceinline__ float sigm(float x) { return 1.0f / (1.0f + __expf(-x)); }

__global__ void word_gate_kernel(const float* __restrict__ GI, const float* __restrict__ GH,
                                 const float* __restrict__ b_ih, const float* __restrict__ b_hh,
                                 float* __restrict__ word_h, bf16* __restrict__ word_h_bf) {
    int idx = blockIdx.x * blockDim.x + threadIdx.x;    // 0 .. 32*512-1
    int m = idx >> 9;
    int j = idx & 511;
    const float* gi = GI + (size_t)m * G3;
    const float* gh = GH + (size_t)m * G3;
    float ir = gi[j]            + b_ih[j];
    float iz = gi[HH + j]       + b_ih[HH + j];
    float in = gi[2 * HH + j]   + b_ih[2 * HH + j];
    float hr = gh[j]            + b_hh[j];
    float hz = gh[HH + j]       + b_hh[HH + j];
    float hn = gh[2 * HH + j]   + b_hh[2 * HH + j];
    float r = sigm(ir + hr);
    float z = sigm(iz + hz);
    float n = tanhf(in + r * hn);
    float h = word_h[idx];
    float hnew = (1.0f - z) * n + z * h;
    word_h[idx]    = hnew;
    word_h_bf[idx] = (bf16)hnew;
}

// sentence GRU + EOS-mask blend of (word_h, sent_h)
__global__ void sent_gate_kernel(const float* __restrict__ GI, const float* __restrict__ GH,
                                 const float* __restrict__ b_ih, const float* __restrict__ b_hh,
                                 const int* __restrict__ targets, int t,
                                 float* __restrict__ word_h, bf16* __restrict__ word_h_bf,
                                 float* __restrict__ sent_h, bf16* __restrict__ sent_h_bf) {
    int idx = blockIdx.x * blockDim.x + threadIdx.x;
    int m = idx >> 9;
    int j = idx & 511;
    const float* gi = GI + (size_t)m * G3;
    const float* gh = GH + (size_t)m * G3;
    float ir = gi[j]          + b_ih[j];
    float iz = gi[HH + j]     + b_ih[HH + j];
    float in = gi[2 * HH + j] + b_ih[2 * HH + j];
    float hr = gh[j]          + b_hh[j];
    float hz = gh[HH + j]     + b_hh[HH + j];
    float hn = gh[2 * HH + j] + b_hh[2 * HH + j];
    float r = sigm(ir + hr);
    float z = sigm(iz + hz);
    float n = tanhf(in + r * hn);
    float sh  = sent_h[idx];
    float tmp = (1.0f - z) * n + z * sh;
    int   nw   = targets[m * SS + t + 1];
    float mask = (nw == EOS_IDX) ? 1.0f : 0.0f;
    float s2 = sh * (1.0f - mask) + tmp * mask;
    float wh = word_h[idx];
    float w2 = wh * (1.0f - mask) + tmp * mask;
    sent_h[idx]    = s2;  sent_h_bf[idx] = (bf16)s2;
    word_h[idx]    = w2;  word_h_bf[idx] = (bf16)w2;
}

// ---------------------------------------------------------------------------
// Per-row log-softmax reduction + masked NLL accumulate (one block per batch)
// ---------------------------------------------------------------------------
__global__ void __launch_bounds__(256)
softmax_loss_kernel(const float* __restrict__ logits,
                    const int* __restrict__ targets,
                    const int* __restrict__ targets_len,
                    int t, float* __restrict__ loss) {
    int b   = blockIdx.x;          // 0..31
    int tid = threadIdx.x;         // 256 threads
    const float* row = logits + (size_t)b * VV;
    __shared__ float red[256];

    float mx = -INFINITY;
    for (int i = tid; i < VV; i += 256) mx = fmaxf(mx, row[i]);
    red[tid] = mx; __syncthreads();
    for (int s = 128; s > 0; s >>= 1) {
        if (tid < s) red[tid] = fmaxf(red[tid], red[tid + s]);
        __syncthreads();
    }
    mx = red[0]; __syncthreads();

    float sum = 0.0f;
    for (int i = tid; i < VV; i += 256) sum += __expf(row[i] - mx);
    red[tid] = sum; __syncthreads();
    for (int s = 128; s > 0; s >>= 1) {
        if (tid < s) red[tid] += red[tid + s];
        __syncthreads();
    }
    if (tid == 0) {
        int   nw    = targets[b * SS + t + 1];
        float lp    = row[nw] - mx - __logf(red[0]);
        float valid = (targets_len[b] > (t + 1)) ? 1.0f : 0.0f;
        atomicAdd(loss, -lp * valid);
    }
}

// ---------------------------------------------------------------------------
// Host orchestration
// ---------------------------------------------------------------------------
extern "C" void kernel_launch(void* const* d_in, const int* in_sizes, int n_in,
                              void* d_out, int out_size, void* d_ws, size_t ws_size,
                              hipStream_t stream) {
    (void)in_sizes; (void)n_in; (void)out_size; (void)ws_size;

    const int*   targets     = (const int*)  d_in[0];
    const int*   targets_kws = (const int*)  d_in[1];
    const float* sent_state  = (const float*)d_in[2];
    const int*   targets_len = (const int*)  d_in[3];
    const float* emb         = (const float*)d_in[4];
    const float* w_ih_w      = (const float*)d_in[5];
    const float* w_hh_w      = (const float*)d_in[6];
    const float* b_ih_w      = (const float*)d_in[7];
    const float* b_hh_w      = (const float*)d_in[8];
    const float* w_ih_s      = (const float*)d_in[9];
    const float* w_hh_s      = (const float*)d_in[10];
    const float* b_ih_s      = (const float*)d_in[11];
    const float* w_hh_s2     = (const float*)d_in[12]; // b_hh_s
    const float* out_w       = (const float*)d_in[13];
    const float* out_b       = (const float*)d_in[14];
    const float* b_hh_s      = w_hh_s2;

    float* loss = (float*)d_out;

    // ---- workspace bump allocator (256B aligned) ----
    char* ws = (char*)d_ws;
    size_t off = 0;
    auto alloc = [&](size_t bytes) -> char* {
        char* p = ws + off;
        off = (off + bytes + 255) & ~(size_t)255;
        return p;
    };
    bf16*  out_w_bf  = (bf16*) alloc((size_t)VV * HH * 2);   // 32.8 MB, L2-resident
    bf16*  wihw_bf   = (bf16*) alloc((size_t)G3 * 2 * EE * 2);
    bf16*  whhw_bf   = (bf16*) alloc((size_t)G3 * HH * 2);
    bf16*  wihs_bf   = (bf16*) alloc((size_t)G3 * HH * 2);
    bf16*  whhs_bf   = (bf16*) alloc((size_t)G3 * HH * 2);
    bf16*  x_bf      = (bf16*) alloc((size_t)BB * 2 * EE * 2);
    float* word_h    = (float*)alloc((size_t)BB * HH * 4);
    float* sent_h    = (float*)alloc((size_t)BB * HH * 4);
    bf16*  word_h_bf = (bf16*) alloc((size_t)BB * HH * 2);
    bf16*  sent_h_bf = (bf16*) alloc((size_t)BB * HH * 2);
    float* GI        = (float*)alloc((size_t)BB * G3 * 4);
    float* GH        = (float*)alloc((size_t)BB * G3 * 4);
    float* logits    = (float*)alloc((size_t)BB * VV * 4);

    // ---- one-time (per launch) weight conversion to bf16 ----
    auto cvt = [&](const float* s, bf16* d, int n) {
        f32_to_bf16_kernel<<<(n + 255) / 256, 256, 0, stream>>>(s, d, n);
    };
    cvt(out_w,  out_w_bf, VV * HH);
    cvt(w_ih_w, wihw_bf,  G3 * 2 * EE);
    cvt(w_hh_w, whhw_bf,  G3 * HH);
    cvt(w_ih_s, wihs_bf,  G3 * HH);
    cvt(w_hh_s, whhs_bf,  G3 * HH);

    init_state_kernel<<<(BB * HH + 255) / 256, 256, 0, stream>>>(
        sent_state, word_h, sent_h, word_h_bf, sent_h_bf, loss);

    // ---- sequential scan over S-1 timesteps ----
    for (int t = 0; t < SS - 1; ++t) {
        // x = concat(emb[kw], emb[lw])  -> bf16
        embed_kernel<<<(BB * 2 * EE + 255) / 256, 256, 0, stream>>>(
            emb, targets, targets_kws, t, x_bf);

        // word GRU: GI = x @ w_ih_w^T (K=1024), GH = h @ w_hh_w^T (K=512)
        gemm32_bf16_kernel<<<G3 / 64, 128, 0, stream>>>(
            x_bf, wihw_bf, nullptr, GI, G3, 2 * EE, 0);
        gemm32_bf16_kernel<<<G3 / 64, 128, 0, stream>>>(
            word_h_bf, whhw_bf, nullptr, GH, G3, HH, 0);
        word_gate_kernel<<<(BB * HH + 255) / 256, 256, 0, stream>>>(
            GI, GH, b_ih_w, b_hh_w, word_h, word_h_bf);

        // vocab projection (dominant): (32x512) x (512x32000) + out_b
        gemm32_bf16_kernel<<<VV / 64, 128, 0, stream>>>(
            word_h_bf, out_w_bf, out_b, logits, VV, HH, 1);

        // log-softmax + masked NLL accumulate
        softmax_loss_kernel<<<BB, 256, 0, stream>>>(
            logits, targets, targets_len, t, loss);

        // sentence GRU on (word_h, sent_h), then EOS-mask blend
        gemm32_bf16_kernel<<<G3 / 64, 128, 0, stream>>>(
            word_h_bf, wihs_bf, nullptr, GI, G3, HH, 0);
        gemm32_bf16_kernel<<<G3 / 64, 128, 0, stream>>>(
            sent_h_bf, whhs_bf, nullptr, GH, G3, HH, 0);
        sent_gate_kernel<<<(BB * HH + 255) / 256, 256, 0, stream>>>(
            GI, GH, b_ih_s, b_hh_s, targets, t,
            word_h, word_h_bf, sent_h, sent_h_bf);
    }
}
